// SpectralPath_49039936585855
// MI455X (gfx1250) — matrix operand order repaired and checked by reading, not compiled, not run on hardware
//
#include <hip/hip_runtime.h>

// ---------------------------------------------------------------------------
// SpectralPath on MI455X (gfx1250): every DFT / projection recast as bf16
// WMMA GEMMs (v_wmma_f32_16x16x32_bf16), f32 accumulate. Memory-bound
// workload (~250MB traffic vs ~7 GFLOP), so intermediates staged in bf16.
// The dominant feats@lin_w GEMM (K=45056) stages tiles into LDS with
// CDNA5 async loads (global_load_async_to_lds_b128 + s_wait_asynccnt) and
// uses split-K for WGP occupancy (deterministic partial-sum reduce).
// ---------------------------------------------------------------------------

typedef __attribute__((ext_vector_type(16))) __bf16       v16bf;
typedef __attribute__((ext_vector_type(8)))  float        v8f;
typedef __attribute__((ext_vector_type(8)))  unsigned int v8u;

#define WMMA_BF16(a, b, acc) \
  __builtin_amdgcn_wmma_f32_16x16x32_bf16(false, (a), false, (b), (short)0, (acc), false, false)

// ---- sizes -----------------------------------------------------------------
#define NS   64        // n*l samples
#define TT   11        // T_TOT
#define NF   5         // FIN
#define NC   32        // LIFT
#define PIX  4096      // 64*64
#define DIM  256
#define DLIN 45056
#define DQ   4096
#define LIN_KSPLIT 8

// ---- WMMA operand loaders (ISA 7.12.2 layouts, wave32) ---------------------
// A (16x32 MxK, bf16): lane&15 = M row; lane>>4 selects K half; element j -> K
//   = (j<8 ? j : j+8) + 8*half.  Pairs (2j,2j+1) are K-contiguous -> dword loads.
__device__ __forceinline__ v16bf load_a16(const __bf16* Ap, int half) {
  v8u au;
#pragma unroll
  for (int jp = 0; jp < 8; ++jp) {
    int k = (jp < 4 ? 2 * jp : 2 * jp + 8) + (half << 3);
    au[jp] = *(const unsigned int*)(Ap + k);
  }
  return __builtin_bit_cast(v16bf, au);
}
// B (32x16 KxN, bf16): lane&15 = N col; element j -> K = (j<8 ? j : j+8) + 8*half (matches A).
__device__ __forceinline__ v16bf load_b16(const __bf16* B, long ldb, long col, int k0, int half) {
  v16bf b;
#pragma unroll
  for (int j = 0; j < 16; ++j)
    b[j] = B[(long)(k0 + (j < 8 ? j : j + 8) + (half << 3)) * ldb + col];
  return b;
}
// K-loop over row-major A (lda) x row-major B (ldb), f32 accumulator tile.
__device__ __forceinline__ v8f wmma_rm(const __bf16* A, long lda, const __bf16* B, long ldb,
                                       int K, long m0, long n0) {
  const int lane = threadIdx.x & 31;
  const int lr = lane & 15, half = lane >> 4;
  v8f acc = {};
  const __bf16* Ap = A + (m0 + lr) * lda;
  for (int k0 = 0; k0 < K; k0 += 32) {
    __builtin_prefetch((const void*)(Ap + k0 + 64), 0, 0);   // global_prefetch
    v16bf a = load_a16(Ap + k0, half);
    v16bf b = load_b16(B, ldb, n0 + lr, k0, half);
    acc = WMMA_BF16(a, b, acc);
  }
  return acc;
}
// CDNA5 async copy: 16B global -> LDS, tracked by ASYNCcnt.
__device__ __forceinline__ void async_g2l_b128(const __bf16* lds_ptr, const void* gaddr) {
  unsigned lds_off = (unsigned)(unsigned long long)lds_ptr;  // low 32b of flat = LDS offset
  asm volatile("global_load_async_to_lds_b128 %0, %1, off"
               :: "v"(lds_off), "v"(gaddr) : "memory");
}
__device__ __forceinline__ void wait_async0() {
  asm volatile("s_wait_asynccnt 0" ::: "memory");
}

// ---- basis generation (bf16), regenerated every launch ---------------------
__global__ void k_basis(__bf16* Wx, __bf16* Wy2, __bf16* Wt2, __bf16* B1, __bf16* B2) {
  int idx = blockIdx.x * blockDim.x + threadIdx.x;
  const float PI2 = 6.283185307179586f;
  if (idx < 1024) {                       // Wx [64][16] fwd x-DFT
    int xw = idx >> 4, col = idx & 15, k = col >> 1, r = col & 1;
    float th = PI2 * (float)(k * xw) / 64.f;
    Wx[idx] = (__bf16)(r ? -__sinf(th) : __cosf(th));
  } else if (idx < 3072) {                // Wy2 [128][16] fwd y-DFT (complex)
    int q = idx - 1024, row = q >> 4, col = q & 15;
    int y = row >> 1, r = row & 1, ky = col >> 1, r2 = col & 1;
    float th = PI2 * (float)(ky * y) / 64.f;
    float c = __cosf(th), s = __sinf(th);
    Wy2[q] = (__bf16)((r == r2) ? c : ((r == 1 && r2 == 0) ? s : -s));
  } else if (idx < 4096) {                // Wt2 [32][32] fwd t-DFT (22 used)
    int q = idx - 3072, row = q >> 5, col = q & 31;
    float v = 0.f;
    if (row < 22 && col < 22) {
      int t = row >> 1, r = row & 1, kt = col >> 1, r2 = col & 1;
      float th = PI2 * (float)(kt * t) / 11.f;
      float c = __cosf(th), s = __sinf(th);
      v = (r == r2) ? c : ((r == 1 && r2 == 0) ? s : -s);
    }
    Wt2[q] = (__bf16)v;
  } else if (idx < 8192) {                // B1 [32][128] inv y (e^{+i})/64
    int q = idx - 4096, row = q >> 7, col = q & 127;
    float v = 0.f;
    if (row < 16) {
      int ky = row >> 1, r = row & 1, py = col >> 1, r2 = col & 1;
      float th = PI2 * (float)(ky * py) / 64.f;
      float c = __cosf(th) * (1.f / 64.f), s = __sinf(th) * (1.f / 64.f);
      v = (r == 0) ? (r2 == 0 ? c : s) : (r2 == 0 ? -s : c);
    }
    B1[q] = (__bf16)v;
  } else if (idx < 10240) {               // B2 [32][64] inv x c2r /64
    int q = idx - 8192, row = q >> 6, px = q & 63;
    float v = 0.f;
    if (row < 16) {
      int kx = row >> 1, r = row & 1;
      float w = (kx == 0) ? 1.f : 2.f;
      float th = PI2 * (float)(kx * px) / 64.f;
      if (r == 0) v = w * __cosf(th) * (1.f / 64.f);
      else        v = (kx == 0) ? 0.f : -w * __sinf(th) * (1.f / 64.f);
    }
    B2[q] = (__bf16)v;
  }
}

// ---- utility kernels -------------------------------------------------------
__global__ void k_f2b(const float* __restrict__ s, __bf16* __restrict__ d, long n) {
  long i = (long)blockIdx.x * blockDim.x + threadIdx.x;
  if (i < n) d[i] = (__bf16)s[i];
}
__global__ void k_zb(__bf16* __restrict__ d, long n) {
  long i = (long)blockIdx.x * blockDim.x + threadIdx.x;
  if (i < n) d[i] = (__bf16)0.f;
}

// ---- stage 1: channel assembly -> A[s][t][f][y][x] bf16 --------------------
__global__ void k_asm(const float* __restrict__ x, __bf16* __restrict__ A) {
  long idx = (long)blockIdx.x * blockDim.x + threadIdx.x;
  if (idx >= (long)NS * TT * NF * PIX) return;
  long p = idx & 4095;
  long r = idx >> 12;                   // s*55 + t*5 + f
  int tf = (int)(r % 55);
  long s = r / 55;
  int t = tf / 5, f = tf % 5;
  int ch = (t < 10) ? (t * 5 + f) : (f < 3 ? 50 + f : f);   // grid_dup: f=3,4 -> ch 3,4
  A[idx] = (__bf16)x[(s * 53 + ch) * PIX + p];
}

// ---- stage 2: x-DFT GEMM, epilogue transposes to C1T[(stf,kx)][(y,r)] ------
__global__ void k_xdft(const __bf16* __restrict__ A, const __bf16* __restrict__ Wx,
                       __bf16* __restrict__ C1T) {
  int wave = (int)(((long)blockIdx.x * blockDim.x + threadIdx.x) >> 5);
  if (wave >= (NS * TT * NF * 64) / 16) return;                 // 14080 tiles
  long m0 = (long)wave << 4;
  v8f acc = wmma_rm(A, 64, Wx, 16, 64, m0, 0);
  const int lane = threadIdx.x & 31, lr = lane & 15, half = lane >> 4;
  int kx = lr >> 1, r = lr & 1;
#pragma unroll
  for (int i = 0; i < 8; ++i) {
    long m = m0 + i + (half << 3);
    long y = m & 63, stf = m >> 6;
    C1T[(stf * 8 + kx) * 128 + y * 2 + r] = (__bf16)acc[i];
  }
}

// ---- stage 3: y-DFT GEMM (complex via K=128), out F2 f32 [st][sp][f][r] ----
__global__ void k_ydft(const __bf16* __restrict__ C1T, const __bf16* __restrict__ Wy2,
                       float* __restrict__ F2) {
  int wave = (int)(((long)blockIdx.x * blockDim.x + threadIdx.x) >> 5);
  if (wave >= (NS * TT * NF * 8) / 16) return;                  // 1760 tiles
  long m0 = (long)wave << 4;
  v8f acc = wmma_rm(C1T, 128, Wy2, 16, 128, m0, 0);
  const int lane = threadIdx.x & 31, lr = lane & 15, half = lane >> 4;
  int ky = lr >> 1, r = lr & 1;
#pragma unroll
  for (int i = 0; i < 8; ++i) {
    long m = m0 + i + (half << 3);
    int kx = (int)(m & 7);
    long stf = m >> 3;
    long f = stf % 5, st = stf / 5;
    F2[((st * 64 + ky * 8 + kx) * 5 + f) * 2 + r] = acc[i];
  }
}

// ---- stage 4: p_w mix (K=5, VALU) -> G bf16 [(s,c,sp)][32pad (t,r)] --------
__global__ void k_pwmix(const float* __restrict__ F2, const float* __restrict__ pw,
                        const float* __restrict__ pb, __bf16* __restrict__ G) {
  long idx = (long)blockIdx.x * blockDim.x + threadIdx.x;
  if (idx >= (long)NS * NC * 64 * 32) return;
  int col = (int)(idx & 31);
  int sp  = (int)((idx >> 5) & 63);
  int c   = (int)((idx >> 11) & 31);
  long s  = idx >> 16;
  float v = 0.f;
  if (col < 22) {
    int t = col >> 1, r = col & 1;
    const float* f2 = F2 + ((s * 11 + t) * 64 + sp) * 10 + r;
#pragma unroll
    for (int f = 0; f < 5; ++f) v += f2[f * 2] * pw[f * 32 + c];
    if (sp == 0 && r == 0) v += 4096.f * pb[c];   // p_b: DC of 64x64 spatial DFT
  }
  G[idx] = (__bf16)v;
}

// ---- stage 5: T-DFT GEMM, epilogue scatters to feats in reference order ----
__global__ void k_tdft(const __bf16* __restrict__ G, const __bf16* __restrict__ Wt2,
                       __bf16* __restrict__ feats) {
  int wave = (int)(((long)blockIdx.x * blockDim.x + threadIdx.x) >> 5);
  if (wave >= ((NS * NC * 64) / 16) * 2) return;                // 16384 tiles
  long m0 = (long)(wave >> 1) << 4;
  long n0 = (long)(wave & 1) << 4;
  v8f acc = wmma_rm(G, 32, Wt2, 32, 32, m0, n0);
  const int lane = threadIdx.x & 31, lr = lane & 15, half = lane >> 4;
  int n = (int)n0 + lr;
  if (n >= 22) return;
  int kt = n >> 1, r = n & 1;
#pragma unroll
  for (int i = 0; i < 8; ++i) {
    long m = m0 + i + (half << 3);
    int sp = (int)(m & 63);
    long sc = m >> 6;
    long c = sc & 31, s = sc >> 5;
    feats[s * DLIN + (long)r * 22528 + (c * 11 + kt) * 64 + sp] = (__bf16)acc[i];
  }
}

// ---- dominant GEMM: feats(64x45056) @ lin_w(45056x256), LDS-staged ---------
// Split-K x8, 64 workgroups x 8 waves.  Per chunk of 64 K:
//   A tile 16x64 (2KB) + B tile 64x128 (16KB) staged via async b128 loads,
//   s_wait_asynccnt 0 + barrier, then 2 WMMA k-steps per wave from LDS.
__global__ void __launch_bounds__(256)
k_lin_lds(const __bf16* __restrict__ A, const __bf16* __restrict__ B,
          float* __restrict__ Part) {
  __shared__ __bf16 sA[16 * 64];
  __shared__ __bf16 sB[64 * 128];
  if (__builtin_amdgcn_cluster_id_x() > 64) return;   // uniform, never taken
  const int tid = threadIdx.x;
  const int wg  = blockIdx.x;                 // 0..63
  const int mn  = wg & 7;                     // mtile(4) x ngroup(2)
  const int ks  = wg >> 3;                    // k-split slice 0..7
  const long m0  = (long)(mn >> 1) << 4;
  const long n0g = (long)(mn & 1) << 7;
  const int wv  = tid >> 5;                   // wave -> local ntile 0..7
  const int lane = tid & 31, lr = lane & 15, half = lane >> 4;

  v8f acc = {};
  const int KCH = DLIN / LIN_KSPLIT;          // 5632 (88 chunks of 64)
  const long kbase = (long)ks * KCH;
  for (int kc = 0; kc < KCH; kc += 64) {
    long k0 = kbase + kc;
    if (tid < 128) {                          // stage A: 128 x b128
      int row = tid >> 3, kk = (tid & 7) * 8;
      async_g2l_b128(&sA[row * 64 + kk], (const void*)(A + (m0 + row) * DLIN + k0 + kk));
    }
#pragma unroll
    for (int j = 0; j < 4; ++j) {             // stage B: 1024 x b128
      int idx = tid + j * 256;
      int kr = idx >> 4, cc = (idx & 15) * 8;
      async_g2l_b128(&sB[kr * 128 + cc], (const void*)(B + (k0 + kr) * 256 + n0g + cc));
    }
    wait_async0();
    __syncthreads();
#pragma unroll
    for (int kk = 0; kk < 64; kk += 32) {     // consume from LDS (ds_load)
      v16bf a = load_a16(&sA[lr * 64 + kk], half);
      v16bf b = load_b16(&sB[kk * 128], 128, wv * 16 + lr, 0, half);
      acc = WMMA_BF16(a, b, acc);
    }
    __syncthreads();
  }
  long n = n0g + wv * 16 + lr;
#pragma unroll
  for (int i = 0; i < 8; ++i) {
    long m = m0 + i + (half << 3);
    Part[((long)ks * NS + m) * DIM + n] = acc[i];
  }
}
__global__ void k_lin_reduce(const float* __restrict__ Part, const float* __restrict__ bias,
                             const float* __restrict__ pos, float* __restrict__ h) {
  int idx = blockIdx.x * blockDim.x + threadIdx.x;
  if (idx >= NS * DIM) return;
  int n = idx & 255, m = idx >> 8;
  float v = bias[n] + pos[(m & 15) * DIM + n];
#pragma unroll
  for (int s = 0; s < LIN_KSPLIT; ++s) v += Part[((long)s * NS + m) * DIM + n];
  h[idx] = v;
}

// ---- generic bf16 GEMM: C = A@B (+bias)(+resid)(relu)(bf16 mirror) ---------
__global__ void k_gemm(const __bf16* __restrict__ A, int lda,
                       const __bf16* __restrict__ B, int ldb,
                       float* __restrict__ C, int ldc,
                       const float* __restrict__ bias,
                       const float* __restrict__ resid,
                       __bf16* __restrict__ Cb,
                       int relu, int M, int N, int K) {
  int wave = (int)(((long)blockIdx.x * blockDim.x + threadIdx.x) >> 5);
  int nt = N >> 4;
  if (wave >= (M >> 4) * nt) return;
  long m0 = (long)(wave / nt) << 4;
  long n0 = (long)(wave % nt) << 4;
  v8f acc = wmma_rm(A, lda, B, ldb, K, m0, n0);
  const int lane = threadIdx.x & 31, lr = lane & 15, half = lane >> 4;
  long n = n0 + lr;
#pragma unroll
  for (int i = 0; i < 8; ++i) {
    long m = m0 + i + (half << 3);
    float v = acc[i];
    if (bias)  v += bias[n];
    if (resid) v += resid[m * ldc + n];
    if (relu)  v = v > 0.f ? v : 0.f;
    if (C)  C[m * ldc + n] = v;
    if (Cb) Cb[m * ldc + n] = (__bf16)v;
  }
}

// ---- LayerNorm (per 256-d token) -> bf16 ----------------------------------
__global__ void k_ln(const float* __restrict__ X, const float* __restrict__ g,
                     const float* __restrict__ b, __bf16* __restrict__ Y) {
  __shared__ float red[DIM];
  int t = threadIdx.x;
  long row = blockIdx.x;
  float v = X[row * DIM + t];
  red[t] = v; __syncthreads();
  for (int s = DIM / 2; s > 0; s >>= 1) { if (t < s) red[t] += red[t + s]; __syncthreads(); }
  float mean = red[0] * (1.f / DIM); __syncthreads();
  float d = v - mean;
  red[t] = d * d; __syncthreads();
  for (int s = DIM / 2; s > 0; s >>= 1) { if (t < s) red[t] += red[t + s]; __syncthreads(); }
  float var = red[0] * (1.f / DIM);
  Y[row * DIM + t] = (__bf16)(d * rsqrtf(var + 1e-5f) * g[t] + b[t]);
}

// ---- attention (NH=1, L=16, hd=256): one block per batch -------------------
__global__ void k_attn(const float* __restrict__ qkv, __bf16* __restrict__ O) {
  __shared__ float sq[16][DIM], sk[16][DIM], sv[16][DIM], sa[16][16];
  int b = blockIdx.x, t = threadIdx.x;
  for (int e = t; e < 16 * DIM; e += 256) {
    int i = e >> 8, d = e & 255;
    long base = (long)(b * 16 + i) * (3 * DIM);
    sq[i][d] = qkv[base + d];
    sk[i][d] = qkv[base + DIM + d];
    sv[i][d] = qkv[base + 2 * DIM + d];
  }
  __syncthreads();
  int i = t >> 4, j = t & 15;
  float s = 0.f;
  for (int d = 0; d < DIM; ++d) s += sq[i][d] * sk[j][d];
  s *= (1.f / 16.f);                      // 1/sqrt(hd=256)
  sa[i][j] = s; __syncthreads();
  float mx = -1e30f;
  for (int jj = 0; jj < 16; ++jj) mx = fmaxf(mx, sa[i][jj]);
  float e = __expf(s - mx), sum = 0.f;
  for (int jj = 0; jj < 16; ++jj) sum += __expf(sa[i][jj] - mx);
  float a = e / sum;
  __syncthreads(); sa[i][j] = a; __syncthreads();
  for (int d = j; d < DIM; d += 16) {
    float o = 0.f;
    for (int jj = 0; jj < 16; ++jj) o += sa[i][jj] * sv[jj][d];
    O[(long)(b * 16 + i) * DIM + d] = (__bf16)o;
  }
}

// ---- inverse stage 1: ky -> py (K=32, 16 used) -----------------------------
__global__ void k_i1(const __bf16* __restrict__ O, const __bf16* __restrict__ B1,
                     __bf16* __restrict__ H) {
  int wave = (int)(((long)blockIdx.x * blockDim.x + threadIdx.x) >> 5);
  if (wave >= (16384 / 16) * 8) return;
  long m0 = (long)(wave / 8) << 4;
  long n0 = (long)(wave % 8) << 4;
  const int lane = threadIdx.x & 31, lr = lane & 15, half = lane >> 4;
  long m = m0 + lr;                       // row = (s*32+c)*8 + kx
  int kx = (int)(m & 7);
  long sc = m >> 3;
  v16bf a;
#pragma unroll
  for (int j = 0; j < 16; ++j) {
    int k = (j < 8 ? j : j + 8) + (half << 3);
    __bf16 v = (__bf16)0.f;
    if (k < 16) {
      int ky = k >> 1, r = k & 1;
      v = O[(sc >> 5) * DQ + (long)r * 2048 + (sc & 31) * 64 + ky * 8 + kx];
    }
    a[j] = v;
  }
  v16bf b = load_b16(B1, 128, n0 + lr, 0, half);
  v8f acc = {};
  acc = WMMA_BF16(a, b, acc);
  int n = (int)n0 + lr;
  int py = n >> 1, r2 = n & 1;
#pragma unroll
  for (int i = 0; i < 8; ++i) {
    long mm = m0 + i + (half << 3);
    int kxx = (int)(mm & 7);
    long scc = mm >> 3;
    H[(scc * 64 + py) * 32 + kxx * 2 + r2] = (__bf16)acc[i];
  }
}

// ---- inverse stage 2: kx -> px (c2r), out Y[(s,pp)][c] ---------------------
__global__ void k_i2(const __bf16* __restrict__ H, const __bf16* __restrict__ B2,
                     float* __restrict__ Y) {
  int wave = (int)(((long)blockIdx.x * blockDim.x + threadIdx.x) >> 5);
  if (wave >= (131072 / 16) * 4) return;
  long m0 = (long)(wave / 4) << 4;
  long n0 = (long)(wave % 4) << 4;
  v8f acc = wmma_rm(H, 32, B2, 64, 32, m0, n0);
  const int lane = threadIdx.x & 31, lr = lane & 15, half = lane >> 4;
  long n = n0 + lr;                       // px
#pragma unroll
  for (int i = 0; i < 8; ++i) {
    long m = m0 + i + (half << 3);        // (s*32+c)*64 + py
    long py = m & 63, sc = m >> 6;
    long s = sc >> 5, c = sc & 31;
    Y[(s * PIX + py * 64 + n) * 32 + c] = acc[i];
  }
}

// ---- down projection (K=32, N=3) + final layout ----------------------------
__global__ void k_down(const float* __restrict__ Y, const float* __restrict__ dw,
                       const float* __restrict__ db, float* __restrict__ out) {
  long idx = (long)blockIdx.x * blockDim.x + threadIdx.x;
  if (idx >= (long)NS * PIX) return;
  long s = idx >> 12, pp = idx & 4095;
  const float* yp = Y + idx * 32;
  float o0 = db[0], o1 = db[1], o2 = db[2];
#pragma unroll
  for (int c = 0; c < 32; ++c) {
    float v = yp[c];
    o0 += v * dw[c * 3 + 0];
    o1 += v * dw[c * 3 + 1];
    o2 += v * dw[c * 3 + 2];
  }
  out[(s * 3 + 0) * PIX + pp] = o0;
  out[(s * 3 + 1) * PIX + pp] = o1;
  out[(s * 3 + 2) * PIX + pp] = o2;
}

// ---------------------------------------------------------------------------
extern "C" void kernel_launch(void* const* d_in, const int* in_sizes, int n_in,
                              void* d_out, int out_size, void* d_ws, size_t ws_size,
                              hipStream_t stream) {
  (void)in_sizes; (void)n_in; (void)out_size; (void)ws_size;
  const float* x      = (const float*)d_in[0];
  const float* p_w    = (const float*)d_in[1];
  const float* p_b    = (const float*)d_in[2];
  const float* lin_w  = (const float*)d_in[3];
  const float* lin_b  = (const float*)d_in[4];
  const float* q_w    = (const float*)d_in[5];
  const float* q_b    = (const float*)d_in[6];
  const float* down_w = (const float*)d_in[7];
  const float* down_b = (const float*)d_in[8];
  const float* pos    = (const float*)d_in[9];
  struct Lp { const float *ln1_g,*ln1_b,*wqkv,*bqkv,*wo,*bo,*ln2_g,*ln2_b,*w1,*b1,*w2,*b2; } L[2];
  for (int l = 0; l < 2; ++l) {
    int o = 10 + 12 * l;
    L[l].ln1_g = (const float*)d_in[o+0];  L[l].ln1_b = (const float*)d_in[o+1];
    L[l].wqkv  = (const float*)d_in[o+2];  L[l].bqkv  = (const float*)d_in[o+3];
    L[l].wo    = (const float*)d_in[o+4];  L[l].bo    = (const float*)d_in[o+5];
    L[l].ln2_g = (const float*)d_in[o+6];  L[l].ln2_b = (const float*)d_in[o+7];
    L[l].w1    = (const float*)d_in[o+8];  L[l].b1    = (const float*)d_in[o+9];
    L[l].w2    = (const float*)d_in[o+10]; L[l].b2    = (const float*)d_in[o+11];
  }

  char* ws = (char*)d_ws;
  size_t off = 0;
  auto alloc = [&](size_t bytes) -> char* {
    char* p = ws + off;
    off += (bytes + 255) & ~(size_t)255;
    return p;
  };

  __bf16* Wx    = (__bf16*)alloc(1024 * 2);
  __bf16* Wy2   = (__bf16*)alloc(2048 * 2);
  __bf16* Wt2   = (__bf16*)alloc(1024 * 2);
  __bf16* B1    = (__bf16*)alloc(4096 * 2);
  __bf16* B2    = (__bf16*)alloc(2048 * 2);
  __bf16* linwb = (__bf16*)alloc((size_t)DLIN * DIM * 2);
  __bf16* qwb   = (__bf16*)alloc((size_t)DIM * DQ * 2);
  __bf16* wqkvb[2], *wob[2], *w1b[2], *w2b[2];
  for (int l = 0; l < 2; ++l) {
    wqkvb[l] = (__bf16*)alloc((size_t)DIM * 3 * DIM * 2);
    wob[l]   = (__bf16*)alloc((size_t)DIM * DIM * 2);
    w1b[l]   = (__bf16*)alloc((size_t)DIM * DIM * 2);
    w2b[l]   = (__bf16*)alloc((size_t)DIM * DIM * 2);
  }
  // big region: assembly A + C1T, later reused for Y (both dead by then)
  char*   big   = alloc(36045056);
  __bf16* Aasm  = (__bf16*)big;                    // 28,835,840 B
  __bf16* C1T   = (__bf16*)(big + 28835840);       //  7,208,960 B
  float*  Yb    = (float*)big;                     // 33,554,432 B (reuse)
  float*  F2    = (float*)alloc((size_t)450560 * 4);
  __bf16* G     = (__bf16*)alloc((size_t)4194304 * 2);   // reused as H later
  __bf16* Hbuf  = G;
  __bf16* feats = (__bf16*)alloc((size_t)NS * DLIN * 2);
  float*  Part  = (float*)alloc((size_t)LIN_KSPLIT * NS * DIM * 4);
  float*  h     = (float*)alloc((size_t)NS * DIM * 4);
  __bf16* xln   = (__bf16*)alloc((size_t)NS * DIM * 2);
  float*  qkvf  = (float*)alloc((size_t)NS * 3 * DIM * 4);
  __bf16* attno = (__bf16*)alloc((size_t)NS * DIM * 2);
  float*  m1f   = (float*)alloc((size_t)NS * DIM * 4);
  __bf16* m1b   = (__bf16*)alloc((size_t)NS * DIM * 2);
  __bf16* hb    = (__bf16*)alloc((size_t)NS * DIM * 2);
  float*  ofel  = (float*)alloc((size_t)NS * DQ * 4);
  __bf16* ofeb  = (__bf16*)alloc((size_t)NS * DQ * 2);

  // ---- basis + bf16 weight staging ----
  hipLaunchKernelGGL(k_basis, dim3(40), dim3(256), 0, stream, Wx, Wy2, Wt2, B1, B2);
  hipLaunchKernelGGL(k_f2b, dim3(45056), dim3(256), 0, stream, lin_w, linwb, (long)DLIN * DIM);
  hipLaunchKernelGGL(k_f2b, dim3(4096),  dim3(256), 0, stream, q_w, qwb, (long)DIM * DQ);
  for (int l = 0; l < 2; ++l) {
    hipLaunchKernelGGL(k_f2b, dim3(768), dim3(256), 0, stream, L[l].wqkv, wqkvb[l], (long)DIM * 3 * DIM);
    hipLaunchKernelGGL(k_f2b, dim3(256), dim3(256), 0, stream, L[l].wo, wob[l], (long)DIM * DIM);
    hipLaunchKernelGGL(k_f2b, dim3(256), dim3(256), 0, stream, L[l].w1, w1b[l], (long)DIM * DIM);
    hipLaunchKernelGGL(k_f2b, dim3(256), dim3(256), 0, stream, L[l].w2, w2b[l], (long)DIM * DIM);
  }

  // ---- forward spectral path ----
  hipLaunchKernelGGL(k_asm,   dim3(56320), dim3(256), 0, stream, x, Aasm);
  hipLaunchKernelGGL(k_xdft,  dim3(3520),  dim3(128), 0, stream, Aasm, Wx, C1T);
  hipLaunchKernelGGL(k_ydft,  dim3(440),   dim3(128), 0, stream, C1T, Wy2, F2);
  hipLaunchKernelGGL(k_pwmix, dim3(16384), dim3(256), 0, stream, F2, p_w, p_b, G);
  hipLaunchKernelGGL(k_tdft,  dim3(4096),  dim3(128), 0, stream, G, Wt2, feats);

  // ---- h = feats @ lin_w + lin_b + pos_embed (async-LDS, split-K) ----
  hipLaunchKernelGGL(k_lin_lds,    dim3(64), dim3(256), 0, stream, feats, linwb, Part);
  hipLaunchKernelGGL(k_lin_reduce, dim3(64), dim3(256), 0, stream, Part, lin_b, pos, h);

  // ---- transformer x2 ----
  for (int l = 0; l < 2; ++l) {
    hipLaunchKernelGGL(k_ln, dim3(NS), dim3(DIM), 0, stream, h, L[l].ln1_g, L[l].ln1_b, xln);
    hipLaunchKernelGGL(k_gemm, dim3(48), dim3(128), 0, stream,
                       xln, DIM, wqkvb[l], 3 * DIM, qkvf, 3 * DIM,
                       L[l].bqkv, (const float*)nullptr, (__bf16*)nullptr, 0,
                       NS, 3 * DIM, DIM);
    hipLaunchKernelGGL(k_attn, dim3(4), dim3(256), 0, stream, qkvf, attno);
    hipLaunchKernelGGL(k_gemm, dim3(16), dim3(128), 0, stream,
                       attno, DIM, wob[l], DIM, h, DIM,
                       L[l].bo, h, (__bf16*)nullptr, 0, NS, DIM, DIM);
    hipLaunchKernelGGL(k_ln, dim3(NS), dim3(DIM), 0, stream, h, L[l].ln2_g, L[l].ln2_b, xln);
    hipLaunchKernelGGL(k_gemm, dim3(16), dim3(128), 0, stream,
                       xln, DIM, w1b[l], DIM, m1f, DIM,
                       L[l].b1, (const float*)nullptr, m1b, 1, NS, DIM, DIM);
    hipLaunchKernelGGL(k_gemm, dim3(16), dim3(128), 0, stream,
                       m1b, DIM, w2b[l], DIM, h, DIM,
                       L[l].b2, h, (__bf16*)nullptr, 0, NS, DIM, DIM);
  }

  // ---- o = h @ q_w + q_b ----
  hipLaunchKernelGGL(k_f2b, dim3(64), dim3(256), 0, stream, h, hb, (long)NS * DIM);
  hipLaunchKernelGGL(k_gemm, dim3(256), dim3(128), 0, stream,
                     hb, DIM, qwb, DQ, ofel, DQ,
                     q_b, (const float*)nullptr, ofeb, 0, NS, DQ, DIM);

  // ---- inverse spectral path + down projection ----
  hipLaunchKernelGGL(k_zb,   dim3(16384), dim3(256), 0, stream, Hbuf, (long)4194304);
  hipLaunchKernelGGL(k_i1,   dim3(2048),  dim3(128), 0, stream, ofeb, B1, Hbuf);
  hipLaunchKernelGGL(k_i2,   dim3(8192),  dim3(128), 0, stream, Hbuf, B2, Yb);
  hipLaunchKernelGGL(k_down, dim3(1024),  dim3(256), 0, stream, Yb, down_w, down_b, (float*)d_out);
}
